// FIBERCrossModalFusion_34978213658700
// MI455X (gfx1250) — matrix operand order repaired and checked by reading, not compiled, not run on hardware
//
#include <hip/hip_runtime.h>
#include <hip/hip_bf16.h>
#include <math.h>

// ---------------------------------------------------------------------------
// FIBER cross-modal fusion for MI455X (gfx1250): wave32, WMMA f32_16x16x32_f16
// ---------------------------------------------------------------------------

typedef __attribute__((ext_vector_type(16))) _Float16 v16h;
typedef __attribute__((ext_vector_type(8)))  float    v8f;

#define BATCH 16
#define SEQ   512
#define TOK   513          // SEQ + 1 (vision token prepended)
#define HDIM  768
#define FFDIM 3072
#define NHEAD 12
#define HD    64
#define NLAYER 3
#define SCPAD 544          // 34 * 16, padded token count for attention tiles
#define LDK   48           // padded LDS row stride (f16): 96B, 32B-aligned, ~2-way banks

// ---------------------------------------------------------------------------
// 1. per-weight-matrix abs-mean scale  s = clip(mean|W|, 1e-5, 1e3)
// ---------------------------------------------------------------------------
struct ScaleArgs { const float* W[32]; int n[32]; };

__global__ __launch_bounds__(256)
void scale_kernel(ScaleArgs args, float* __restrict__ scales) {
    const int m = blockIdx.x;
    const float* W = args.W[m];
    const int n = args.n[m];
    __shared__ float red[256];
    float s = 0.f;
    for (int i = threadIdx.x; i < n; i += 256) s += fabsf(W[i]);
    red[threadIdx.x] = s;
    __syncthreads();
    for (int off = 128; off > 0; off >>= 1) {
        if (threadIdx.x < off) red[threadIdx.x] += red[threadIdx.x + off];
        __syncthreads();
    }
    if (threadIdx.x == 0) {
        float mean = red[0] / (float)n;
        scales[m] = fminf(fmaxf(mean, 1e-5f), 1000.0f);
    }
}

// ---------------------------------------------------------------------------
// 2. Tiled WMMA GEMM:  C[M,N] = act(A[M,K] @ ternarize(W[N,K])^T + bias) (+res)
//    BM=BN=128, BK=32, 256 threads = 8 waves, each wave 2x4 WMMA tiles.
//    Software-pipelined, double-buffered LDS; weights ternarized on the fly
//    (branchless: cmp + cndmask + copysign).  K % 32 == 0 in this model.
// ---------------------------------------------------------------------------
__global__ __launch_bounds__(256)
void gemm_bit_wmma(const float* __restrict__ A, const float* __restrict__ W,
                   const float* __restrict__ bias, const float* residual,
                   const float* __restrict__ s_ptr, const float* __restrict__ temp_ptr,
                   float* C, int M, int N, int K, int gelu)
{
    __shared__ _Float16 As[2][128 * LDK];
    __shared__ _Float16 Bs[2][128 * LDK];

    const int tid  = threadIdx.x;
    const int lane = tid & 31;
    const int wave = tid >> 5;
    const int m0 = blockIdx.y * 128;
    const int n0 = blockIdx.x * 128;

    const bool quant = (s_ptr != nullptr);
    float s  = 1.0f, th = 0.0f;
    if (quant) { s = s_ptr[0]; th = s * (2.0f / 3.0f); }

    // wave computes 32x64 sub-tile: rows m0+wr.., cols n0+wc..
    const int wr = (wave >> 1) * 32;
    const int wc = (wave & 1) * 64;

    v8f acc[2][4] = {};

    // staging: thread covers row = tid/2 (0..127), half-row of 16 cols
    const int srow = tid >> 1;
    const int scol = (tid & 1) * 16;
    const int nk = K >> 5;

    // clamped row pointers: loads always legal, invalid rows zeroed via select
    const bool aok = (m0 + srow) < M;
    const bool bok = (n0 + srow) < N;
    const float* arow = &A[(size_t)(aok ? (m0 + srow) : 0) * K + scol];
    const float* brow = &W[(size_t)(bok ? (n0 + srow) : 0) * K + scol];

    float va[16], vb[16];

    auto gload = [&](int kt) {
        const float* ap = arow + kt * 32;
        const float* bp = brow + kt * 32;
#pragma unroll
        for (int j = 0; j < 4; ++j) {
            const float4 fa = *(const float4*)(ap + j * 4);
            const float4 fb = *(const float4*)(bp + j * 4);
            va[j * 4 + 0] = aok ? fa.x : 0.f; va[j * 4 + 1] = aok ? fa.y : 0.f;
            va[j * 4 + 2] = aok ? fa.z : 0.f; va[j * 4 + 3] = aok ? fa.w : 0.f;
            vb[j * 4 + 0] = bok ? fb.x : 0.f; vb[j * 4 + 1] = bok ? fb.y : 0.f;
            vb[j * 4 + 2] = bok ? fb.z : 0.f; vb[j * 4 + 3] = bok ? fb.w : 0.f;
        }
        // speculative prefetch of following K tile (drop-safe per ISA 10.5)
        __builtin_prefetch(ap + 32, 0, 1);
        __builtin_prefetch(bp + 32, 0, 1);
    };

    auto sstore = [&](int buf) {
        v16h ha, hb;
        if (quant) {            // uniform scalar branch
#pragma unroll
            for (int j = 0; j < 16; ++j) {
                ha[j] = (_Float16)va[j];
                const float w = vb[j];
                // branchless ternarize: v_cmp + v_cndmask + sign transfer
                const float mag = (fabsf(w) > th) ? s : 0.0f;
                hb[j] = (_Float16)__builtin_copysignf(mag, w);
            }
        } else {
#pragma unroll
            for (int j = 0; j < 16; ++j) {
                ha[j] = (_Float16)va[j];
                hb[j] = (_Float16)vb[j];
            }
        }
        *(v16h*)&As[buf][srow * LDK + scol] = ha;   // 2x ds_store_b128
        *(v16h*)&Bs[buf][srow * LDK + scol] = hb;
    };

    auto compute = [&](int buf) {
        v16h af[2], bf[4];
#pragma unroll
        for (int i = 0; i < 2; ++i)
            af[i] = *(const v16h*)&As[buf][(wr + i * 16 + (lane & 15)) * LDK + (lane >> 4) * 16];
#pragma unroll
        for (int j = 0; j < 4; ++j)
            bf[j] = *(const v16h*)&Bs[buf][(wc + j * 16 + (lane & 15)) * LDK + (lane >> 4) * 16];
#pragma unroll
        for (int i = 0; i < 2; ++i)
#pragma unroll
            for (int j = 0; j < 4; ++j)
                acc[i][j] = __builtin_amdgcn_wmma_f32_16x16x32_f16(
                    false, af[i], false, bf[j], (short)0, acc[i][j], false, false);
    };

    // software pipeline: load(k+1) || wmma(k), one barrier per iteration
    gload(0);
    sstore(0);
    __syncthreads();
    for (int kt = 0; kt < nk; ++kt) {
        const int cur = kt & 1;
        if (kt + 1 < nk) gload(kt + 1);
        compute(cur);
        if (kt + 1 < nk) sstore(cur ^ 1);
        __syncthreads();
    }

    // ---- epilogue: bias / gelu / residual / recip-scale ----
    const float oscale = temp_ptr ? (1.0f / temp_ptr[0]) : 1.0f;
    const bool interior = (m0 + 128 <= M) && (n0 + 128 <= N);

    auto epilogue = [&](bool guard) {
#pragma unroll
        for (int i = 0; i < 2; ++i) {
#pragma unroll
            for (int j = 0; j < 4; ++j) {
                const int col = n0 + wc + j * 16 + (lane & 15);
                if (guard && col >= N) continue;
                const float bb = bias ? bias[col] : 0.0f;
#pragma unroll
                for (int e = 0; e < 8; ++e) {
                    const int row = m0 + wr + i * 16 + e + (lane >> 4) * 8;
                    if (guard && row >= M) continue;
                    float x = acc[i][j][e] + bb;
                    if (gelu) x = 0.5f * x * (1.0f + erff(x * 0.70710678f));
                    if (residual) x += residual[(size_t)row * N + col];
                    C[(size_t)row * N + col] = x * oscale;
                }
            }
        }
    };
    if (interior) epilogue(false);    // uniform branch: guard-free stores
    else          epilogue(true);
}

// ---------------------------------------------------------------------------
// 3. WMMA attention: one wave per (b, h, 16-row q tile).
//    scores (16 x 544) -> masked softmax in LDS -> ctx (16 x 64)
// ---------------------------------------------------------------------------
__global__ __launch_bounds__(32)
void attn_kernel(const float* __restrict__ Q, const float* __restrict__ Kc,
                 const float* __restrict__ V, const float* __restrict__ mask,
                 float* __restrict__ ctx)
{
    __shared__ float    sc[16 * SCPAD];
    __shared__ _Float16 P[16 * SCPAD];

    const int lane = threadIdx.x;
    const int qt = blockIdx.x;     // 0..32
    const int h  = blockIdx.y;     // 0..11
    const int b  = blockIdx.z;     // 0..15
    const int q0 = qt * 16;
    const float scale = 0.125f;    // 1/sqrt(64)
    const size_t base = ((size_t)b * TOK) * HDIM + (size_t)h * HD;

    // ---- load Q fragments (two K-halves of the 64-dim head) ----
    v16h qf[2];
    {
        const int tok = q0 + (lane & 15);
#pragma unroll
        for (int s2 = 0; s2 < 2; ++s2) {
            v16h a = {};
            if (tok < TOK) {
                const float* p = &Q[base + (size_t)tok * HDIM + s2 * 32 + (lane >> 4) * 16];
#pragma unroll
                for (int j = 0; j < 4; ++j) {
                    const float4 f = *(const float4*)(p + j * 4);
                    a[j * 4 + 0] = (_Float16)f.x; a[j * 4 + 1] = (_Float16)f.y;
                    a[j * 4 + 2] = (_Float16)f.z; a[j * 4 + 3] = (_Float16)f.w;
                }
            }
            qf[s2] = a;
        }
    }

    // ---- scores = Q K^T * scale + additive mask ----
    for (int kt = 0; kt < SCPAD / 16; ++kt) {
        v8f acc = {};
        const int tokb = kt * 16;
        const int tok = tokb + (lane & 15);
        const bool full = (tokb + 16 <= TOK);   // uniform: fast path for all but the last two tiles
#pragma unroll
        for (int s2 = 0; s2 < 2; ++s2) {
            v16h bf = {};
            if (full || tok < TOK) {
                const float* p = &Kc[base + (size_t)tok * HDIM + s2 * 32 + (lane >> 4) * 16];
#pragma unroll
                for (int j = 0; j < 4; ++j) {
                    const float4 f = *(const float4*)(p + j * 4);
                    bf[j * 4 + 0] = (_Float16)f.x; bf[j * 4 + 1] = (_Float16)f.y;
                    bf[j * 4 + 2] = (_Float16)f.z; bf[j * 4 + 3] = (_Float16)f.w;
                }
            }
            acc = __builtin_amdgcn_wmma_f32_16x16x32_f16(
                false, qf[s2], false, bf, (short)0, acc, false, false);
        }
#pragma unroll
        for (int e = 0; e < 8; ++e) {
            const int r = e + (lane >> 4) * 8;       // 0..15
            const int c = tokb + (lane & 15);
            float x = -1e30f;
            if (c < TOK && (q0 + r) < TOK) {
                const float m = (c == 0) ? 1.0f : mask[b * SEQ + (c - 1)];
                x = acc[e] * scale + (1.0f - m) * -10000.0f;
            }
            sc[r * SCPAD + c] = x;
        }
    }
    __syncthreads();

    // ---- softmax over 544 (padded cols are -1e30 -> exp == 0) ----
    if (lane < 16) {
        const int r = lane;
        float mx = -1e30f;
        for (int c = 0; c < SCPAD; ++c) mx = fmaxf(mx, sc[r * SCPAD + c]);
        float sum = 0.f;
        for (int c = 0; c < SCPAD; ++c) {
            const float e = __expf(sc[r * SCPAD + c] - mx);
            sc[r * SCPAD + c] = e;
            sum += e;
        }
        const float inv = 1.0f / sum;
        for (int c = 0; c < SCPAD; ++c) P[r * SCPAD + c] = (_Float16)(sc[r * SCPAD + c] * inv);
    }
    __syncthreads();

    // ---- ctx = P @ V : 4 col-tiles x 17 K-steps of 32 ----
#pragma unroll
    for (int n = 0; n < 4; ++n) {
        v8f acc = {};
        for (int kt = 0; kt < SCPAD / 32; ++kt) {
            const v16h af = *(const v16h*)&P[(lane & 15) * SCPAD + kt * 32 + (lane >> 4) * 16];
            v16h bf = {};
            const int tok0 = kt * 32 + (lane >> 4) * 16;
            const bool full = (kt * 32 + 32 <= TOK);
#pragma unroll
            for (int j = 0; j < 16; ++j) {
                const int tok = tok0 + j;
                if (full || tok < TOK)
                    bf[j] = (_Float16)V[base + (size_t)tok * HDIM + n * 16 + (lane & 15)];
            }
            acc = __builtin_amdgcn_wmma_f32_16x16x32_f16(
                false, af, false, bf, (short)0, acc, false, false);
        }
#pragma unroll
        for (int e = 0; e < 8; ++e) {
            const int r = q0 + e + (lane >> 4) * 8;
            const int c = n * 16 + (lane & 15);
            if (r < TOK) ctx[base + (size_t)r * HDIM + c] = acc[e];
        }
    }
}

// ---------------------------------------------------------------------------
// 4. LayerNorm: one wave per row (H=768 -> 24 elems/lane, wave32 shfl reduce)
// ---------------------------------------------------------------------------
__global__ __launch_bounds__(256)
void layernorm_kernel(const float* __restrict__ x, const float* __restrict__ g,
                      const float* __restrict__ be, float* __restrict__ out, int rows)
{
    const int wave = threadIdx.x >> 5;
    const int lane = threadIdx.x & 31;
    const int r = blockIdx.x * 8 + wave;
    if (r >= rows) return;
    const float* xr = x + (size_t)r * HDIM;

    float sum = 0.f;
#pragma unroll
    for (int j = 0; j < 24; ++j) sum += xr[lane + 32 * j];
#pragma unroll
    for (int o = 16; o > 0; o >>= 1) sum += __shfl_xor(sum, o, 32);
    const float mu = sum / (float)HDIM;

    float vs = 0.f;
#pragma unroll
    for (int j = 0; j < 24; ++j) { const float d = xr[lane + 32 * j] - mu; vs += d * d; }
#pragma unroll
    for (int o = 16; o > 0; o >>= 1) vs += __shfl_xor(vs, o, 32);
    const float inv = rsqrtf(vs / (float)HDIM + 1e-5f);

    float* orow = out + (size_t)r * HDIM;
#pragma unroll
    for (int j = 0; j < 24; ++j) {
        const int c = lane + 32 * j;
        orow[c] = (xr[c] - mu) * inv * g[c] + be[c];
    }
}

// ---------------------------------------------------------------------------
// 5. concat / split of [vision ; text] token axis (one block per row: no
//    per-element div/mod, contiguous b32 copies)
// ---------------------------------------------------------------------------
__global__ __launch_bounds__(256)
void merge_kernel(const float* __restrict__ vp, const float* __restrict__ tp,
                  float* __restrict__ comb) {
    const int row = blockIdx.x;          // 0 .. BATCH*TOK-1
    const int b = row / TOK;
    const int t = row - b * TOK;
    const float* src = (t == 0) ? &vp[(size_t)b * HDIM]
                                : &tp[((size_t)b * SEQ + (t - 1)) * HDIM];
    float* dst = &comb[(size_t)row * HDIM];
    for (int c = threadIdx.x; c < HDIM; c += 256) dst[c] = src[c];
}

__global__ __launch_bounds__(256)
void split_kernel(const float* __restrict__ comb, float* __restrict__ v,
                  float* __restrict__ t) {
    const int row = blockIdx.x;          // 0 .. BATCH*TOK-1
    const int b = row / TOK;
    const int tt = row - b * TOK;
    const float* src = &comb[(size_t)row * HDIM];
    float* dst = (tt == 0) ? &v[(size_t)b * HDIM]
                           : &t[((size_t)b * SEQ + (tt - 1)) * HDIM];
    for (int c = threadIdx.x; c < HDIM; c += 256) dst[c] = src[c];
}

// ---------------------------------------------------------------------------
// 6. masked mean pool + L2 normalize
// ---------------------------------------------------------------------------
__global__ __launch_bounds__(256)
void pool_kernel(const float* __restrict__ tfin, const float* __restrict__ mask,
                 float* __restrict__ tpool) {
    const int b = blockIdx.x;
    for (int c = threadIdx.x; c < HDIM; c += blockDim.x) {
        float s = 0.f, ms = 0.f;
        for (int j = 0; j < SEQ; ++j) {
            const float m = mask[b * SEQ + j];
            s += tfin[((size_t)b * SEQ + j) * HDIM + c] * m;
            ms += m;
        }
        tpool[b * HDIM + c] = s / ms;
    }
}

__global__ __launch_bounds__(32)
void l2norm_kernel(const float* __restrict__ x, float* __restrict__ out) {
    const int r = blockIdx.x;
    const int lane = threadIdx.x;
    const float* xr = x + (size_t)r * HDIM;
    float s = 0.f;
#pragma unroll
    for (int j = 0; j < 24; ++j) { const float v = xr[lane + 32 * j]; s += v * v; }
#pragma unroll
    for (int o = 16; o > 0; o >>= 1) s += __shfl_xor(s, o, 32);
    const float inv = 1.0f / fmaxf(sqrtf(s), 1e-12f);
#pragma unroll
    for (int j = 0; j < 24; ++j) {
        const int c = lane + 32 * j;
        out[(size_t)r * HDIM + c] = xr[c] * inv;
    }
}

// ---------------------------------------------------------------------------
// host orchestration
// ---------------------------------------------------------------------------
extern "C" void kernel_launch(void* const* d_in, const int* in_sizes, int n_in,
                              void* d_out, int out_size, void* d_ws, size_t ws_size,
                              hipStream_t stream) {
    // input order (setup_inputs dict order)
    const float* vision = (const float*)d_in[0];
    const float* text   = (const float*)d_in[1];
    const float* mask   = (const float*)d_in[2];
    const float* Wvp = (const float*)d_in[3];  const float* bvp = (const float*)d_in[4];
    const float* Wtp = (const float*)d_in[5];  const float* btp = (const float*)d_in[6];
    const float* Wq  = (const float*)d_in[7];  const float* bq  = (const float*)d_in[8];
    const float* Wk  = (const float*)d_in[9];  const float* bk  = (const float*)d_in[10];
    const float* Wv  = (const float*)d_in[11]; const float* bv  = (const float*)d_in[12];
    const float* Wo  = (const float*)d_in[13]; const float* bo  = (const float*)d_in[14];
    const float* ln_g = (const float*)d_in[15]; const float* ln_b = (const float*)d_in[16];
    const float* vfg = (const float*)d_in[17]; const float* vfb = (const float*)d_in[18];
    const float* vW1 = (const float*)d_in[19]; const float* vb1 = (const float*)d_in[20];
    const float* vW2 = (const float*)d_in[21]; const float* vb2 = (const float*)d_in[22];
    const float* tfg = (const float*)d_in[23]; const float* tfb = (const float*)d_in[24];
    const float* tW1 = (const float*)d_in[25]; const float* tb1 = (const float*)d_in[26];
    const float* tW2 = (const float*)d_in[27]; const float* tb2 = (const float*)d_in[28];
    const float* Wfv = (const float*)d_in[29]; const float* bfv = (const float*)d_in[30];
    const float* Wft = (const float*)d_in[31]; const float* bft = (const float*)d_in[32];
    const float* temp = (const float*)d_in[33];

    // workspace carve-up (f32)
    float* ws = (float*)d_ws;
    size_t off = 0;
    auto alloc = [&](size_t n) { float* p = ws + off; off += (n + 63) & ~(size_t)63; return p; };
    float* scales  = alloc(64);
    float* vbuf    = alloc((size_t)BATCH * HDIM);
    float* tbuf    = alloc((size_t)BATCH * SEQ * HDIM);
    float* vp_tmp  = alloc((size_t)BATCH * HDIM);
    float* tp_tmp  = alloc((size_t)BATCH * SEQ * HDIM);
    float* comb    = alloc((size_t)BATCH * TOK * HDIM);
    float* qb      = alloc((size_t)BATCH * TOK * HDIM);
    float* kb      = alloc((size_t)BATCH * TOK * HDIM);
    float* vlb     = alloc((size_t)BATCH * TOK * HDIM);
    float* ctxb    = alloc((size_t)BATCH * TOK * HDIM);
    float* outb    = alloc((size_t)BATCH * TOK * HDIM);
    float* out2    = alloc((size_t)BATCH * TOK * HDIM);
    float* h1      = alloc((size_t)BATCH * SEQ * HDIM);
    float* h1v     = alloc((size_t)BATCH * HDIM);
    float* hid     = alloc((size_t)BATCH * SEQ * FFDIM);
    float* hidv    = alloc((size_t)BATCH * FFDIM);
    float* tpool   = alloc((size_t)BATCH * HDIM);
    float* vn      = alloc((size_t)BATCH * HDIM);
    float* tn      = alloc((size_t)BATCH * HDIM);

    // output layout: v_fin | t_fin | sim
    float* vfin   = (float*)d_out;
    float* tfin   = vfin + (size_t)BATCH * HDIM;
    float* simout = tfin + (size_t)BATCH * SEQ * HDIM;

    // 1) per-matrix BitNet scales (32 matrices)
    ScaleArgs sa;
    for (int i = 0; i < NLAYER; ++i) {
        const int bse = i * 10;
        sa.W[bse + 0] = Wvp + (size_t)i * HDIM * HDIM;  sa.n[bse + 0] = HDIM * HDIM;
        sa.W[bse + 1] = Wtp + (size_t)i * HDIM * HDIM;  sa.n[bse + 1] = HDIM * HDIM;
        sa.W[bse + 2] = Wq  + (size_t)i * HDIM * HDIM;  sa.n[bse + 2] = HDIM * HDIM;
        sa.W[bse + 3] = Wk  + (size_t)i * HDIM * HDIM;  sa.n[bse + 3] = HDIM * HDIM;
        sa.W[bse + 4] = Wv  + (size_t)i * HDIM * HDIM;  sa.n[bse + 4] = HDIM * HDIM;
        sa.W[bse + 5] = Wo  + (size_t)i * HDIM * HDIM;  sa.n[bse + 5] = HDIM * HDIM;
        sa.W[bse + 6] = vW1 + (size_t)i * FFDIM * HDIM; sa.n[bse + 6] = FFDIM * HDIM;
        sa.W[bse + 7] = vW2 + (size_t)i * HDIM * FFDIM; sa.n[bse + 7] = HDIM * FFDIM;
        sa.W[bse + 8] = tW1 + (size_t)i * FFDIM * HDIM; sa.n[bse + 8] = FFDIM * HDIM;
        sa.W[bse + 9] = tW2 + (size_t)i * HDIM * FFDIM; sa.n[bse + 9] = HDIM * FFDIM;
    }
    sa.W[30] = Wfv; sa.n[30] = HDIM * HDIM;
    sa.W[31] = Wft; sa.n[31] = HDIM * HDIM;
    scale_kernel<<<32, 256, 0, stream>>>(sa, scales);

    // 2) initialize activation streams
    hipMemcpyAsync(vbuf, vision, (size_t)BATCH * HDIM * sizeof(float),
                   hipMemcpyDeviceToDevice, stream);
    hipMemcpyAsync(tbuf, text, (size_t)BATCH * SEQ * HDIM * sizeof(float),
                   hipMemcpyDeviceToDevice, stream);

    auto gemm = [&](const float* A, const float* Wm, const float* bias, const float* res,
                    const float* sp, const float* tp_, float* Cm,
                    int M, int N, int K, int gelu) {
        dim3 grid((N + 127) / 128, (M + 127) / 128);
        gemm_bit_wmma<<<grid, 256, 0, stream>>>(A, Wm, bias, res, sp, tp_, Cm, M, N, K, gelu);
    };

    const int MT = BATCH * TOK;   // 8208
    const int MS = BATCH * SEQ;   // 8192

    for (int i = 0; i < NLAYER; ++i) {
        const size_t wHH = (size_t)i * HDIM * HDIM;
        const size_t wFH = (size_t)i * FFDIM * HDIM;
        const float* sc = scales + i * 10;

        // projections
        gemm(vbuf, Wvp + wHH, bvp + i * HDIM, nullptr, sc + 0, nullptr, vp_tmp, BATCH, HDIM, HDIM, 0);
        gemm(tbuf, Wtp + wHH, btp + i * HDIM, nullptr, sc + 1, nullptr, tp_tmp, MS, HDIM, HDIM, 0);
        merge_kernel<<<MT, 256, 0, stream>>>(vp_tmp, tp_tmp, comb);

        // q, k, v
        gemm(comb, Wq + wHH, bq + i * HDIM, nullptr, sc + 2, nullptr, qb, MT, HDIM, HDIM, 0);
        gemm(comb, Wk + wHH, bk + i * HDIM, nullptr, sc + 3, nullptr, kb, MT, HDIM, HDIM, 0);
        gemm(comb, Wv + wHH, bv + i * HDIM, nullptr, sc + 4, nullptr, vlb, MT, HDIM, HDIM, 0);

        // attention
        attn_kernel<<<dim3(33, NHEAD, BATCH), 32, 0, stream>>>(qb, kb, vlb, mask, ctxb);

        // output projection + residual + LN
        gemm(ctxb, Wo + wHH, bo + i * HDIM, comb, sc + 5, nullptr, outb, MT, HDIM, HDIM, 0);
        layernorm_kernel<<<(MT + 7) / 8, 256, 0, stream>>>(outb, ln_g + i * HDIM, ln_b + i * HDIM, out2, MT);
        split_kernel<<<MT, 256, 0, stream>>>(out2, vbuf, tbuf);

        // vision FFN (pre-LN, GELU, residual)
        layernorm_kernel<<<(BATCH + 7) / 8, 256, 0, stream>>>(vbuf, vfg + i * HDIM, vfb + i * HDIM, h1v, BATCH);
        gemm(h1v, vW1 + wFH, vb1 + i * FFDIM, nullptr, sc + 6, nullptr, hidv, BATCH, FFDIM, HDIM, 1);
        gemm(hidv, vW2 + wFH, vb2 + i * HDIM, vbuf, sc + 7, nullptr, vbuf, BATCH, HDIM, FFDIM, 0);

        // text FFN
        layernorm_kernel<<<(MS + 7) / 8, 256, 0, stream>>>(tbuf, tfg + i * HDIM, tfb + i * HDIM, h1, MS);
        gemm(h1, tW1 + wFH, tb1 + i * FFDIM, nullptr, sc + 8, nullptr, hid, MS, FFDIM, HDIM, 1);
        gemm(hid, tW2 + wFH, tb2 + i * HDIM, tbuf, sc + 9, nullptr, tbuf, MS, HDIM, FFDIM, 0);
    }

    // final projections (straight into d_out)
    gemm(vbuf, Wfv, bfv, nullptr, scales + 30, nullptr, vfin, BATCH, HDIM, HDIM, 0);
    gemm(tbuf, Wft, bft, nullptr, scales + 31, nullptr, tfin, MS, HDIM, HDIM, 0);

    // masked pool, l2 normalize, similarity = (vn @ tn^T) / temp  (un-quantized GEMM)
    pool_kernel<<<BATCH, 256, 0, stream>>>(tfin, mask, tpool);
    l2norm_kernel<<<BATCH, 32, 0, stream>>>(vfin, vn);
    l2norm_kernel<<<BATCH, 32, 0, stream>>>(tpool, tn);
    gemm(vn, tn, nullptr, nullptr, nullptr, temp, simout, BATCH, BATCH, HDIM, 0);
}
